// DimeNetPlusPlusWrap_24163486008148
// MI455X (gfx1250) — matrix operand order repaired and checked by reading, not compiled, not run on hardware
//
#include <hip/hip_runtime.h>
#include <math.h>

// ---------------------------------------------------------------------------
// DimeNet++ forward for MI455X (gfx1250), wave32 + WMMA bf16.
// ---------------------------------------------------------------------------

typedef __attribute__((ext_vector_type(16))) __bf16 v16bf;
typedef __attribute__((ext_vector_type(8)))  float  v8f;

struct U8 { unsigned int d[8]; };
struct alignas(16) U4 { unsigned int d[4]; };

__device__ __forceinline__ unsigned short f2bf(float f) {
  union { float f; unsigned int u; } c; c.f = f;
  unsigned int u = c.u;
  unsigned int r = u + 0x7FFFu + ((u >> 16) & 1u);   // round-to-nearest-even
  return (unsigned short)(r >> 16);
}

__device__ __forceinline__ unsigned long long pack4bf(float4 v) {
  return (unsigned long long)f2bf(v.x) |
         ((unsigned long long)f2bf(v.y) << 16) |
         ((unsigned long long)f2bf(v.z) << 32) |
         ((unsigned long long)f2bf(v.w) << 48);
}

__device__ __forceinline__ float silu(float v) {
  return v * (1.f / (1.f + __expf(-v)));
}

// Load a 16-element bf16 fragment as two 16-byte LDS reads.
__device__ __forceinline__ v16bf load_frag(const unsigned short* base, int dw0, int dw1) {
  const unsigned int* p = (const unsigned int*)base;
  U4 a = *(const U4*)(p + dw0);
  U4 b = *(const U4*)(p + dw1);
  U8 u;
  u.d[0] = a.d[0]; u.d[1] = a.d[1]; u.d[2] = a.d[2]; u.d[3] = a.d[3];
  u.d[4] = b.d[0]; u.d[5] = b.d[1]; u.d[6] = b.d[2]; u.d[7] = b.d[3];
  return __builtin_bit_cast(v16bf, u);
}

// gfx1250 async global->LDS copy (ASYNCcnt-tracked), ISA ch.10 / 08_async_tensor.
__device__ __forceinline__ void async_copy_f32(const float* gsrc, float* ldst) {
  unsigned int lo = (unsigned int)(unsigned long long)(uintptr_t)ldst; // LDS byte addr
  asm volatile("global_load_async_to_lds_b32 %0, %1, off"
               :: "v"(lo), "v"((unsigned long long)(uintptr_t)gsrc)
               : "memory");
}
__device__ __forceinline__ void async_wait0() {
  asm volatile("s_wait_asynccnt 0x0" ::: "memory");
}

// LDS pitches (ushorts). 40*2=80B and 136*2=272B are 16B-aligned; 20/68-dword
// strides keep rows on distinct bank groups.
#define APITCH 40
#define TPITCH 136

// ---------------------------------------------------------------------------
// Tiled bf16 WMMA GEMM:  out[M,N] = epilogue(A[M,K] @ B[K,N])
// epilogue: (+bias[n]) -> (silu) -> (*mulp[m,n]) -> (+resp[m,n])
// 256 threads (8 waves), block tile 128 x TN, K tiles of 32, LDS double-
// buffered with register-staged prefetch of the next tile.
// Safe for out==mulp or out==resp (each element read-then-written by 1 lane).
// ---------------------------------------------------------------------------
template<int TN>
__global__ __launch_bounds__(256) void gemm_bf16_wmma_kernel(
    const float* __restrict__ A, const float* __restrict__ B,
    const float* __restrict__ bias, const float* __restrict__ mulp,
    const float* __restrict__ resp, float* __restrict__ out,
    int M, int N, int K, int do_act)
{
  constexpr int NT = TN / 16;    // 16x16 tiles per wave along N
  constexpr int BV = TN / 32;    // B float4 loads per thread
  __shared__ alignas(16) unsigned short As[2][128 * APITCH];  // [row][k]
  __shared__ alignas(16) unsigned short Bs[2][TN * APITCH];   // [col][k]

  const int tid  = threadIdx.x;
  const int lane = tid & 31;
  const int w    = tid >> 5;
  const int l16  = lane & 15;
  const int hf   = lane >> 4;
  const int mb0  = blockIdx.x * 128;
  const int nb0  = blockIdx.y * TN;

  v8f acc[NT];
#pragma unroll
  for (int t = 0; t < NT; ++t)
#pragma unroll
    for (int r = 0; r < 8; ++r) acc[t][r] = 0.f;

  auto loadA = [&](int k0, float4* ra) {
#pragma unroll
    for (int j = 0; j < 4; ++j) {
      int vi = tid + j * 256;
      int r = vi >> 3, c = (vi & 7) << 2;
      int gm = mb0 + r;
      ra[j] = (gm < M) ? *(const float4*)(A + (size_t)gm * K + k0 + c)
                       : make_float4(0.f, 0.f, 0.f, 0.f);
    }
  };
  auto loadB = [&](int k0, float4* rb) {
#pragma unroll
    for (int j = 0; j < BV; ++j) {
      int vi = tid + j * 256;
      int r = vi / (TN / 4), c = (vi % (TN / 4)) << 2;
      rb[j] = *(const float4*)(B + (size_t)(k0 + r) * N + nb0 + c);
    }
  };
  auto storeA = [&](int p, const float4* ra) {
#pragma unroll
    for (int j = 0; j < 4; ++j) {
      int vi = tid + j * 256;
      int r = vi >> 3, c = (vi & 7) << 2;
      *(unsigned long long*)(&As[p][r * APITCH + c]) = pack4bf(ra[j]);
    }
  };
  auto storeB = [&](int p, const float4* rb) {
#pragma unroll
    for (int j = 0; j < BV; ++j) {
      int vi = tid + j * 256;
      int r = vi / (TN / 4), c = (vi % (TN / 4)) << 2;
      Bs[p][(c + 0) * APITCH + r] = f2bf(rb[j].x);
      Bs[p][(c + 1) * APITCH + r] = f2bf(rb[j].y);
      Bs[p][(c + 2) * APITCH + r] = f2bf(rb[j].z);
      Bs[p][(c + 3) * APITCH + r] = f2bf(rb[j].w);
    }
  };

  { float4 ra[4], rb[4]; loadA(0, ra); loadB(0, rb); storeA(0, ra); storeB(0, rb); }

  const int KT = K >> 5;
  int p = 0;
  for (int kt = 0; kt < KT; ++kt) {
    float4 ra[4], rb[4];
    const bool more = (kt + 1) < KT;
    if (more) { loadA((kt + 1) << 5, ra); loadB((kt + 1) << 5, rb); }
    __syncthreads();
    const unsigned short* Ab = As[p];
    const unsigned short* Bb = Bs[p];
    v16bf af = load_frag(&Ab[(w * 16 + l16) * APITCH], hf * 4, 8 + hf * 4);
    v16bf bfr[NT];
#pragma unroll
    for (int t = 0; t < NT; ++t)
      bfr[t] = load_frag(&Bb[(t * 16 + l16) * APITCH], hf * 4, 8 + hf * 4);
#pragma unroll
    for (int t = 0; t < NT; ++t)
      acc[t] = __builtin_amdgcn_wmma_f32_16x16x32_bf16(
          false, af, false, bfr[t], (short)0, acc[t], false, false);
    if (more) { storeA(p ^ 1, ra); storeB(p ^ 1, rb); }
    p ^= 1;
  }

#pragma unroll
  for (int t = 0; t < NT; ++t) {
    int n = nb0 + t * 16 + l16;
#pragma unroll
    for (int r = 0; r < 8; ++r) {
      int m = mb0 + w * 16 + hf * 8 + r;
      if (m < M) {
        float v = acc[t][r];
        if (bias)   v += bias[n];
        if (do_act) v = silu(v);
        size_t o = (size_t)m * N + n;
        if (mulp) v *= mulp[o];
        if (resp) v += resp[o];
        out[o] = v;
      }
    }
  }
}

// ---------------------------------------------------------------------------
// Dual-weight GEMM (N=K=128): shares one streamed A (xe) between two weights.
//   out0 = silu(A@B0 + bias0)                (x_ji)
//   out1 = silu(A@B1 + bias1) * mul1         (x_kj, in-place mul OK)
// ---------------------------------------------------------------------------
__global__ __launch_bounds__(256) void gemm_dual_bf16_wmma_kernel(
    const float* __restrict__ A,
    const float* __restrict__ B0, const float* __restrict__ bias0,
    const float* __restrict__ B1, const float* __restrict__ bias1,
    float* __restrict__ out0,
    const float* __restrict__ mul1, float* __restrict__ out1, int M)
{
  __shared__ alignas(16) unsigned short As[2][128 * APITCH];
  __shared__ alignas(16) unsigned short Bs0[2][128 * APITCH];
  __shared__ alignas(16) unsigned short Bs1[2][128 * APITCH];

  const int tid  = threadIdx.x;
  const int lane = tid & 31;
  const int w    = tid >> 5;
  const int l16  = lane & 15;
  const int hf   = lane >> 4;
  const int mb0  = blockIdx.x * 128;

  v8f acc0[8], acc1[8];
#pragma unroll
  for (int t = 0; t < 8; ++t)
#pragma unroll
    for (int r = 0; r < 8; ++r) { acc0[t][r] = 0.f; acc1[t][r] = 0.f; }

  auto loadA = [&](int k0, float4* ra) {
#pragma unroll
    for (int j = 0; j < 4; ++j) {
      int vi = tid + j * 256;
      int r = vi >> 3, c = (vi & 7) << 2;
      int gm = mb0 + r;
      ra[j] = (gm < M) ? *(const float4*)(A + (size_t)gm * 128 + k0 + c)
                       : make_float4(0.f, 0.f, 0.f, 0.f);
    }
  };
  auto loadW = [&](const float* B, int k0, float4* rb) {
#pragma unroll
    for (int j = 0; j < 4; ++j) {
      int vi = tid + j * 256;
      int r = vi >> 5, c = (vi & 31) << 2;
      rb[j] = *(const float4*)(B + (size_t)(k0 + r) * 128 + c);
    }
  };
  auto storeA = [&](int p, const float4* ra) {
#pragma unroll
    for (int j = 0; j < 4; ++j) {
      int vi = tid + j * 256;
      int r = vi >> 3, c = (vi & 7) << 2;
      *(unsigned long long*)(&As[p][r * APITCH + c]) = pack4bf(ra[j]);
    }
  };
  auto storeW = [&](unsigned short* Bsp, const float4* rb) {
#pragma unroll
    for (int j = 0; j < 4; ++j) {
      int vi = tid + j * 256;
      int r = vi >> 5, c = (vi & 31) << 2;
      Bsp[(c + 0) * APITCH + r] = f2bf(rb[j].x);
      Bsp[(c + 1) * APITCH + r] = f2bf(rb[j].y);
      Bsp[(c + 2) * APITCH + r] = f2bf(rb[j].z);
      Bsp[(c + 3) * APITCH + r] = f2bf(rb[j].w);
    }
  };

  { float4 ra[4], r0[4], r1[4];
    loadA(0, ra); loadW(B0, 0, r0); loadW(B1, 0, r1);
    storeA(0, ra); storeW(Bs0[0], r0); storeW(Bs1[0], r1); }

  int p = 0;
  for (int kt = 0; kt < 4; ++kt) {
    float4 ra[4], r0[4], r1[4];
    const bool more = kt < 3;
    if (more) { loadA((kt + 1) << 5, ra); loadW(B0, (kt + 1) << 5, r0); loadW(B1, (kt + 1) << 5, r1); }
    __syncthreads();
    v16bf af = load_frag(&As[p][(w * 16 + l16) * APITCH], hf * 4, 8 + hf * 4);
    {
      v16bf bfr[8];
#pragma unroll
      for (int t = 0; t < 8; ++t)
        bfr[t] = load_frag(&Bs0[p][(t * 16 + l16) * APITCH], hf * 4, 8 + hf * 4);
#pragma unroll
      for (int t = 0; t < 8; ++t)
        acc0[t] = __builtin_amdgcn_wmma_f32_16x16x32_bf16(
            false, af, false, bfr[t], (short)0, acc0[t], false, false);
    }
    {
      v16bf bfr[8];
#pragma unroll
      for (int t = 0; t < 8; ++t)
        bfr[t] = load_frag(&Bs1[p][(t * 16 + l16) * APITCH], hf * 4, 8 + hf * 4);
#pragma unroll
      for (int t = 0; t < 8; ++t)
        acc1[t] = __builtin_amdgcn_wmma_f32_16x16x32_bf16(
            false, af, false, bfr[t], (short)0, acc1[t], false, false);
    }
    if (more) { storeA(p ^ 1, ra); storeW(Bs0[p ^ 1], r0); storeW(Bs1[p ^ 1], r1); }
    p ^= 1;
  }

#pragma unroll
  for (int t = 0; t < 8; ++t) {
    int n = t * 16 + l16;
#pragma unroll
    for (int r = 0; r < 8; ++r) {
      int m = mb0 + w * 16 + hf * 8 + r;
      if (m < M) {
        size_t o = (size_t)m * 128 + n;
        out0[o] = silu(acc0[t][r] + bias0[n]);
        out1[o] = silu(acc1[t][r] + bias1[n]) * mul1[o];
      }
    }
  }
}

// ---------------------------------------------------------------------------
// Fused residual MLP (H=128):  out = A + silu(silu(A@W0+b0)@W1+b1)
// t1 lives in LDS (bf16); A tiles register-prefetched. out may alias A.
// ---------------------------------------------------------------------------
__global__ __launch_bounds__(256) void res_fused_kernel(
    const float* __restrict__ A,
    const float* __restrict__ W0, const float* __restrict__ b0,
    const float* __restrict__ W1, const float* __restrict__ b1,
    float* __restrict__ out, int M)
{
  __shared__ alignas(16) unsigned short As[128 * APITCH];
  __shared__ alignas(16) unsigned short Bs[128 * APITCH];
  __shared__ alignas(16) unsigned short T1s[128 * TPITCH];

  const int tid  = threadIdx.x;
  const int lane = tid & 31;
  const int w    = tid >> 5;
  const int l16  = lane & 15;
  const int hf   = lane >> 4;
  const int mb0  = blockIdx.x * 128;

  v8f acc[8];
#pragma unroll
  for (int t = 0; t < 8; ++t)
#pragma unroll
    for (int r = 0; r < 8; ++r) acc[t][r] = 0.f;

  auto loadA4 = [&](int k0, float4* ra) {
#pragma unroll
    for (int j = 0; j < 4; ++j) {
      int vi = tid + j * 256;
      int r = vi >> 3, c = (vi & 7) << 2;
      int gm = mb0 + r;
      ra[j] = (gm < M) ? *(const float4*)(A + (size_t)gm * 128 + k0 + c)
                       : make_float4(0.f, 0.f, 0.f, 0.f);
    }
  };
  auto storeA4 = [&](const float4* ra) {
#pragma unroll
    for (int j = 0; j < 4; ++j) {
      int vi = tid + j * 256;
      int r = vi >> 3, c = (vi & 7) << 2;
      *(unsigned long long*)(&As[r * APITCH + c]) = pack4bf(ra[j]);
    }
  };
  auto stageW = [&](const float* W, int k0) {
#pragma unroll
    for (int j = 0; j < 4; ++j) {
      int vi = tid + j * 256;
      int r = vi >> 5, c = (vi & 31) << 2;
      float4 v = *(const float4*)(W + (size_t)(k0 + r) * 128 + c);
      Bs[(c + 0) * APITCH + r] = f2bf(v.x);
      Bs[(c + 1) * APITCH + r] = f2bf(v.y);
      Bs[(c + 2) * APITCH + r] = f2bf(v.z);
      Bs[(c + 3) * APITCH + r] = f2bf(v.w);
    }
  };

  // ---- Stage 1: t1 = silu(A @ W0 + b0) -> LDS ----
  float4 ra[4];
  loadA4(0, ra);
  for (int kt = 0; kt < 4; ++kt) {
    const int k0 = kt << 5;
    __syncthreads();               // previous compute finished reading LDS
    storeA4(ra);
    stageW(W0, k0);
    // warm L2 for stage 2's weight tile
    __builtin_prefetch((const void*)(W1 + (size_t)k0 * 128 + tid * 16), 0, 2);
    float4 rn[4];
    if (kt < 3) loadA4(k0 + 32, rn);
    __syncthreads();

    v16bf af = load_frag(&As[(w * 16 + l16) * APITCH], hf * 4, 8 + hf * 4);
    v16bf bfr[8];
#pragma unroll
    for (int t = 0; t < 8; ++t)
      bfr[t] = load_frag(&Bs[(t * 16 + l16) * APITCH], hf * 4, 8 + hf * 4);
#pragma unroll
    for (int t = 0; t < 8; ++t)
      acc[t] = __builtin_amdgcn_wmma_f32_16x16x32_bf16(
          false, af, false, bfr[t], (short)0, acc[t], false, false);
    if (kt < 3) { ra[0] = rn[0]; ra[1] = rn[1]; ra[2] = rn[2]; ra[3] = rn[3]; }
  }

  // Epilogue 1: silu(+bias) -> T1s[row][col] (bf16), reset accumulators.
#pragma unroll
  for (int t = 0; t < 8; ++t) {
    int n = t * 16 + l16;
#pragma unroll
    for (int r = 0; r < 8; ++r) {
      int rl = w * 16 + hf * 8 + r;
      T1s[rl * TPITCH + n] = f2bf(silu(acc[t][r] + b0[n]));
      acc[t][r] = 0.f;
    }
  }

  // ---- Stage 2: out = A + silu(t1 @ W1 + b1) ----
  for (int k0 = 0; k0 < 128; k0 += 32) {
    __syncthreads();   // also orders T1s writes before first fragment reads
    stageW(W1, k0);
    __syncthreads();

    v16bf af = load_frag(&T1s[(w * 16 + l16) * TPITCH],
                         (k0 >> 1) + hf * 4, (k0 >> 1) + 8 + hf * 4);
    v16bf bfr[8];
#pragma unroll
    for (int t = 0; t < 8; ++t)
      bfr[t] = load_frag(&Bs[(t * 16 + l16) * APITCH], hf * 4, 8 + hf * 4);
#pragma unroll
    for (int t = 0; t < 8; ++t)
      acc[t] = __builtin_amdgcn_wmma_f32_16x16x32_bf16(
          false, af, false, bfr[t], (short)0, acc[t], false, false);
  }

#pragma unroll
  for (int t = 0; t < 8; ++t) {
    int n = t * 16 + l16;
#pragma unroll
    for (int r = 0; r < 8; ++r) {
      int m = mb0 + w * 16 + hf * 8 + r;
      if (m < M) {
        size_t o = (size_t)m * 128 + n;
        out[o] = A[o] + silu(acc[t][r] + b1[n]);
      }
    }
  }
}

// ---------------------------------------------------------------------------
// rbf2[e,h] = sum_j (sum_r rbf[e,r]*W1[r,j]) * W2[j,h]   (6 -> 8 -> 128)
// ---------------------------------------------------------------------------
__global__ __launch_bounds__(128) void rbf_expand_kernel(
    const float* __restrict__ rbf, const float* __restrict__ W1,
    const float* __restrict__ W2, float* __restrict__ out, int E)
{
  __shared__ float sW1[48];
  __shared__ float sW2[8 * 128];
  __shared__ float sr[6];
  int tid = threadIdx.x;
  int e = blockIdx.x;
  if (tid < 48) sW1[tid] = W1[tid];
  for (int i = tid; i < 1024; i += 128) sW2[i] = W2[i];
  if (tid < 6) sr[tid] = rbf[(size_t)e * 6 + tid];
  __syncthreads();
  float acc = 0.f;
#pragma unroll
  for (int j = 0; j < 8; ++j) {
    float t = 0.f;
#pragma unroll
    for (int r = 0; r < 6; ++r) t += sr[r] * sW1[r * 8 + j];
    acc += t * sW2[j * 128 + tid];
  }
  out[(size_t)e * 128 + tid] = acc;
}

// ---------------------------------------------------------------------------
// Fused triplet kernel: for each triplet t,
//   tmp[j] = sum_s sbf[t,s]*Wsbf1[s,j]                 (42 -> 8, in LDS)
//   agg[idx_ji[t], i] += xkjd[idx_kj[t], i] * (sum_j tmp[j]*Wsbf2[j,i])
// 4 triplets x 64 lanes per block; weights staged via async global->LDS.
// ---------------------------------------------------------------------------
__global__ __launch_bounds__(256) void triplet_fused_kernel(
    const float* __restrict__ sbf, const float* __restrict__ W1,
    const float* __restrict__ W2, const float* __restrict__ xkjd,
    const int* __restrict__ idx_kj, const int* __restrict__ idx_ji,
    float* __restrict__ agg, int T)
{
  __shared__ float sW1[336];
  __shared__ float sW2[512];
  __shared__ float srow[4][42];
  __shared__ float stmp[4][8];
  int tid = threadIdx.x;
  for (int i = tid; i < 336; i += 256) async_copy_f32(&W1[i], &sW1[i]);
  for (int i = tid; i < 512; i += 256) async_copy_f32(&W2[i], &sW2[i]);
  int tq = tid >> 6, i = tid & 63;
  int t = blockIdx.x * 4 + tq;
  bool valid = (t < T);
  if (valid && i < 42) srow[tq][i] = sbf[(size_t)t * 42 + i];
  async_wait0();
  __syncthreads();
  if (valid && i < 8) {
    float a = 0.f;
#pragma unroll
    for (int s = 0; s < 42; ++s) a += srow[tq][s] * sW1[s * 8 + i];
    stmp[tq][i] = a;
  }
  __syncthreads();
  if (!valid) return;
  int kj = idx_kj[t];
  int ji = idx_ji[t];
  float s = 0.f;
#pragma unroll
  for (int j = 0; j < 8; ++j) s += stmp[tq][j] * sW2[j * 64 + i];
  float v = xkjd[(size_t)kj * 64 + i] * s;
  atomicAdd(&agg[(size_t)ji * 64 + i], v);
}

// ---------------------------------------------------------------------------
// nodeT[idx_i[e], h] += (sum_r rbf[e,r]*Worbf[r,h]) * xe[e,h]
// ---------------------------------------------------------------------------
__global__ __launch_bounds__(128) void edge_out_scatter_kernel(
    const float* __restrict__ rbf, const float* __restrict__ Worbf,
    const float* __restrict__ xe, const int* __restrict__ idx_i,
    float* __restrict__ nodeT, int E)
{
  __shared__ float sW[768];
  __shared__ float sr[6];
  int tid = threadIdx.x;
  int e = blockIdx.x;
#pragma unroll
  for (int j = 0; j < 6; ++j)
    async_copy_f32(&Worbf[tid + j * 128], &sW[tid + j * 128]);
  if (tid < 6) sr[tid] = rbf[(size_t)e * 6 + tid];
  async_wait0();
  __syncthreads();
  float v = 0.f;
#pragma unroll
  for (int r = 0; r < 6; ++r) v += sr[r] * sW[r * 128 + tid];
  v *= xe[(size_t)e * 128 + tid];
  atomicAdd(&nodeT[(size_t)idx_i[e] * 128 + tid], v);
}

// ---------------------------------------------------------------------------
// out[n] += sum_k t3[n,k]*Wo[k]   (k = 256, block per node)
// ---------------------------------------------------------------------------
__global__ __launch_bounds__(256) void node_final_kernel(
    const float* __restrict__ t3, const float* __restrict__ Wo,
    float* __restrict__ out, int Nn)
{
  int n = blockIdx.x, tid = threadIdx.x;
  float v = t3[(size_t)n * 256 + tid] * Wo[tid];
#pragma unroll
  for (int off = 16; off > 0; off >>= 1) v += __shfl_down(v, off, 32);
  __shared__ float red[8];
  if ((tid & 31) == 0) red[tid >> 5] = v;
  __syncthreads();
  if (tid == 0) {
    float s = 0.f;
#pragma unroll
    for (int i = 0; i < 8; ++i) s += red[i];
    out[n] += s;
  }
}

__global__ void zero_kernel(float* __restrict__ p, size_t n) {
  size_t i = (size_t)blockIdx.x * blockDim.x + threadIdx.x;
  size_t stride = (size_t)gridDim.x * blockDim.x;
  for (; i < n; i += stride) p[i] = 0.f;
}

// ---------------------------------------------------------------------------
extern "C" void kernel_launch(void* const* d_in, const int* in_sizes, int n_in,
                              void* d_out, int out_size, void* d_ws, size_t ws_size,
                              hipStream_t stream) {
  const float* x     = (const float*)d_in[0];
  const float* rbf   = (const float*)d_in[1];
  const float* sbf   = (const float*)d_in[2];
  const float* Wrbf1 = (const float*)d_in[3];
  const float* Wrbf2 = (const float*)d_in[4];
  const float* Wsbf1 = (const float*)d_in[5];
  const float* Wsbf2 = (const float*)d_in[6];
  const float* Wkj   = (const float*)d_in[7];
  const float* bkj   = (const float*)d_in[8];
  const float* Wji   = (const float*)d_in[9];
  const float* bji   = (const float*)d_in[10];
  const float* Wdown = (const float*)d_in[11];
  const float* Wup   = (const float*)d_in[12];
  const float* Wres  = (const float*)d_in[13];
  const float* bres  = (const float*)d_in[14];
  const float* Wlin  = (const float*)d_in[15];
  const float* blin  = (const float*)d_in[16];
  const float* Worbf = (const float*)d_in[17];
  const float* Woup  = (const float*)d_in[18];
  const float* boup  = (const float*)d_in[19];
  const float* Wouts = (const float*)d_in[20];
  const float* bouts = (const float*)d_in[21];
  const float* Wo    = (const float*)d_in[22];
  const int* idx_kj  = (const int*)d_in[23];
  const int* idx_ji  = (const int*)d_in[24];
  const int* idx_i   = (const int*)d_in[25];

  const int H = 128, IE = 64, OEc = 256;
  const int E  = in_sizes[0] / H;
  const int T  = in_sizes[23];
  const int Nn = out_size;          // OC == 1
  float* outp  = (float*)d_out;

  // Workspace layout (floats)
  float* ws    = (float*)d_ws;
  size_t off   = 0;
  auto wsa = [&](size_t n) { float* p = ws + off; off += n; return p; };
  float* Xbuf  = wsa((size_t)E * H);
  float* Abuf  = wsa((size_t)E * H);
  float* Bbuf  = wsa((size_t)E * H);
  float* xkjd  = wsa((size_t)E * IE);
  float* agg   = wsa((size_t)E * IE);
  float* nodeT = wsa((size_t)Nn * H);
  float* nodeA = wsa((size_t)Nn * OEc);
  float* nodeB = wsa((size_t)Nn * OEc);
  (void)ws_size; (void)n_in;

  auto zero = [&](float* p, size_t n) {
    zero_kernel<<<dim3(2048), dim3(256), 0, stream>>>(p, n);
  };

  auto gemm = [&](const float* A, const float* B, const float* bias,
                  const float* mulp, const float* resp, float* out,
                  int M, int N, int K, int act) {
    dim3 blk(256);
    if (N % 128 == 0) {
      dim3 grd((M + 127) / 128, N / 128);
      gemm_bf16_wmma_kernel<128><<<grd, blk, 0, stream>>>(A, B, bias, mulp, resp, out, M, N, K, act);
    } else {
      dim3 grd((M + 127) / 128, N / 64);
      gemm_bf16_wmma_kernel<64><<<grd, blk, 0, stream>>>(A, B, bias, mulp, resp, out, M, N, K, act);
    }
  };

  auto out_block = [&](const float* xe, int ob) {
    zero(nodeT, (size_t)Nn * H);
    edge_out_scatter_kernel<<<E, 128, 0, stream>>>(
        rbf, Worbf + (size_t)ob * 6 * H, xe, idx_i, nodeT, E);
    gemm(nodeT, Woup + (size_t)ob * H * OEc, boup + (size_t)ob * OEc,
         nullptr, nullptr, nodeA, Nn, OEc, H, 0);
    float* c0 = nodeA; float* c1 = nodeB;
    for (int l = 0; l < 3; ++l) {
      gemm(c0, Wouts + ((size_t)ob * 3 + l) * OEc * OEc,
           bouts + ((size_t)ob * 3 + l) * OEc, nullptr, nullptr, c1,
           Nn, OEc, OEc, 1);
      float* tmp = c0; c0 = c1; c1 = tmp;
    }
    node_final_kernel<<<Nn, 256, 0, stream>>>(c0, Wo + (size_t)ob * OEc, outp, Nn);
  };

  auto res_layer = [&](float* hbuf, int b, int rl) {
    const float* W0 = Wres + (size_t)((b * 3 + rl) * 2 + 0) * H * H;
    const float* b0 = bres + (size_t)((b * 3 + rl) * 2 + 0) * H;
    const float* W1 = Wres + (size_t)((b * 3 + rl) * 2 + 1) * H * H;
    const float* b1 = bres + (size_t)((b * 3 + rl) * 2 + 1) * H;
    dim3 grd((E + 127) / 128, 1);
    res_fused_kernel<<<grd, dim3(256), 0, stream>>>(hbuf, W0, b0, W1, b1, hbuf, E);
  };

  zero(outp, (size_t)Nn);

  const float* xe = x;
  out_block(xe, 0);

  for (int b = 0; b < 4; ++b) {
    // rbf2 = rbf @ Wrbf1 @ Wrbf2 -> Bbuf
    rbf_expand_kernel<<<E, 128, 0, stream>>>(
        rbf, Wrbf1 + (size_t)b * 48, Wrbf2 + (size_t)b * 8 * H, Bbuf, E);
    // x_ji = silu(xe@Wji+bji) -> Abuf ; x_kj = silu(xe@Wkj+bkj)*rbf2 -> Bbuf
    {
      dim3 grd((E + 127) / 128, 1);
      gemm_dual_bf16_wmma_kernel<<<grd, dim3(256), 0, stream>>>(
          xe, Wji + (size_t)b * H * H, bji + (size_t)b * H,
          Wkj + (size_t)b * H * H, bkj + (size_t)b * H,
          Abuf, Bbuf, Bbuf, E);
    }
    // xkjd = silu(x_kj @ Wdown)
    gemm(Bbuf, Wdown + (size_t)b * H * IE, nullptr, nullptr, nullptr,
         xkjd, E, IE, H, 1);
    // agg = segment_sum(xkjd[idx_kj] * (sbf@Wsbf1@Wsbf2), idx_ji)
    zero(agg, (size_t)E * IE);
    triplet_fused_kernel<<<(T + 3) / 4, 256, 0, stream>>>(
        sbf, Wsbf1 + (size_t)b * 42 * 8, Wsbf2 + (size_t)b * 8 * IE,
        xkjd, idx_kj, idx_ji, agg, T);
    // h = x_ji + silu(agg @ Wup)
    gemm(agg, Wup + (size_t)b * IE * H, nullptr, nullptr, Abuf,
         Bbuf, E, H, IE, 1);
    // residual block 0 (before skip), fused, in place
    res_layer(Bbuf, b, 0);
    // h = silu(h @ Wlin + blin) + xe   -> becomes new xe in Xbuf
    gemm(Bbuf, Wlin + (size_t)b * H * H, blin + (size_t)b * H,
         nullptr, xe, Xbuf, E, H, H, 1);
    // residual blocks 1,2 (after skip), in place on Xbuf
    res_layer(Xbuf, b, 1);
    res_layer(Xbuf, b, 2);
    xe = Xbuf;
    out_block(xe, b + 1);
  }
}
